// GraphStructureAttention_18863496364647
// MI455X (gfx1250) — matrix operand order repaired and checked by reading, not compiled, not run on hardware
//
#include <hip/hip_runtime.h>
#include <hip/hip_bf16.h>

#define N_NODES 50000
#define N_EDGES 800000
#define H 128

typedef __attribute__((ext_vector_type(16))) _Float16 v16h;
typedef __attribute__((ext_vector_type(8)))  _Float16 v8h;
typedef __attribute__((ext_vector_type(4)))  _Float16 h4;
typedef __attribute__((ext_vector_type(8)))  float    v8f;

// ---------------------------------------------------------------------------
// Kernel 1: zero the f32 aggregation buffer (ws is poisoned between runs)
// ---------------------------------------------------------------------------
__global__ void zero_agg_kernel(float4* __restrict__ agg) {
    size_t i = (size_t)blockIdx.x * blockDim.x + threadIdx.x;  // 6.4M floats / 4
    agg[i] = float4{0.f, 0.f, 0.f, 0.f};
}

// ---------------------------------------------------------------------------
// Kernel 2: pre-swizzle W1 (f32 -> f16) into CDNA5 WMMA B-fragment layout.
// Fragment index: [nt(8)][kk(8)][lane(32)] -> v16h (16 halfs per lane).
// B (K x N = 32 x 16) layout: lanes 0-15: N=lane, elems 0-7 = K 0..7, 8-15 = K 16..23
//                             lanes 16-31: N=lane-16, elems 0-7 = K 8..15, 8-15 = K 24..31
// ---------------------------------------------------------------------------
__global__ void prep_w1_kernel(const float* __restrict__ W1, v16h* __restrict__ w1frag) {
    int gid  = blockIdx.x * blockDim.x + threadIdx.x;   // 0..2047
    int nt   = gid >> 8;
    int kk   = (gid >> 5) & 7;
    int lane = gid & 31;
    int n     = nt * 16 + (lane & 15);
    int kbase = kk * 32 + ((lane >> 4) << 3);           // +8 for upper half-wave
    v16h v;
#pragma unroll
    for (int e = 0; e < 16; ++e)
        v[e] = (_Float16)W1[(size_t)(kbase + e + ((e >> 3) << 3)) * H + n];
    w1frag[gid] = v;
}

// ---------------------------------------------------------------------------
// Kernel 3: per-edge MLP on WMMA + f32 atomic scatter.
// 8 waves / block, each wave owns a 16-edge tile (M=16).
// ---------------------------------------------------------------------------
__global__ void __launch_bounds__(256)
edge_mlp_scatter_kernel(const float* __restrict__ x,
                        const int*   __restrict__ ei,
                        const v16h*  __restrict__ w1frag,
                        const float* __restrict__ b1,
                        const float* __restrict__ W2,
                        const float* __restrict__ b2,
                        float*       __restrict__ agg)
{
    // Per-wave A-tile: 16 edges x 256 (concat [x_src | x_tgt]) in f16. 8KB/wave.
    __shared__ __align__(16) _Float16 Atile[8][16][256];

    const int wid  = threadIdx.x >> 5;
    const int lane = threadIdx.x & 31;
    const int ebase = (blockIdx.x * 8 + wid) * 16;

    const int* src = ei;             // edge_index[0]
    const int* tgt = ei + N_EDGES;   // edge_index[1]

    // lanes 0-15 carry src ids of edges 0..15; lanes 16-31 carry tgt ids
    int sv = (lane < 16) ? src[ebase + lane] : tgt[ebase + (lane - 16)];

    // ---- Gather [x_src | x_tgt] rows (f32, coalesced 512B/row) into LDS f16
#pragma unroll 4
    for (int e = 0; e < 16; ++e) {
        int s = __shfl(sv, e, 32);
        int t = __shfl(sv, 16 + e, 32);
        float4 xs = *(const float4*)(x + (size_t)s * H + lane * 4);
        float4 xt = *(const float4*)(x + (size_t)t * H + lane * 4);
        h4 hs, ht;
        hs[0] = (_Float16)xs.x; hs[1] = (_Float16)xs.y;
        hs[2] = (_Float16)xs.z; hs[3] = (_Float16)xs.w;
        ht[0] = (_Float16)xt.x; ht[1] = (_Float16)xt.y;
        ht[2] = (_Float16)xt.z; ht[3] = (_Float16)xt.w;
        *(h4*)&Atile[wid][e][lane * 4]       = hs;   // cols 0..127   = x_src
        *(h4*)&Atile[wid][e][128 + lane * 4] = ht;   // cols 128..255 = x_tgt
    }
    // Intra-wave LDS ordering: tile is private to this wave, DS ops are in-order.
    asm volatile("s_wait_dscnt 0" ::: "memory");

    // ---- Load 8 A fragments (16x32 f16 each) in ISA lane layout:
    // lanes 0-15: row=lane,  elems 0-7 = K 0..7,  8-15 = K 16..23
    // lanes16-31: row=l-16,  elems 0-7 = K 8..15, 8-15 = K 24..31
    const int row = lane & 15;
    const int hb  = (lane >> 4) << 3;   // 0 or 8
    v16h a[8];
#pragma unroll
    for (int kk = 0; kk < 8; ++kk) {
        v8h lo = *(const v8h*)&Atile[wid][row][kk * 32 + hb];
        v8h hi = *(const v8h*)&Atile[wid][row][kk * 32 + 16 + hb];
#pragma unroll
        for (int j = 0; j < 8; ++j) { a[kk][j] = lo[j]; a[kk][j + 8] = hi[j]; }
    }

    // ---- Layer 1 GEMM on WMMA, fused bias+ReLU+layer-2 dot.
    // C layout: lane n = lane&15; acc[r] = h[m][n], m = r (lanes<16) / r+8.
    const int ncol = lane & 15;
    float s2[8] = {0.f, 0.f, 0.f, 0.f, 0.f, 0.f, 0.f, 0.f};
#pragma unroll
    for (int nt = 0; nt < 8; ++nt) {
        v8f acc = {};
        const v16h* bp = w1frag + nt * 256;
#pragma unroll
        for (int kk = 0; kk < 8; ++kk) {
            v16h b = bp[kk * 32 + lane];
            acc = __builtin_amdgcn_wmma_f32_16x16x32_f16(
                false, a[kk], false, b, (short)0, acc, false, false);
        }
        float b1v = b1[nt * 16 + ncol];
        float w2v = W2[nt * 16 + ncol];
#pragma unroll
        for (int r = 0; r < 8; ++r) {
            float hv = fmaxf(acc[r] + b1v, 0.0f);     // bias + ReLU
            s2[r] = fmaf(hv, w2v, s2[r]);             // h @ W2 partial
        }
    }
    // Reduce layer-2 dot over the 16 N-lanes of each half-wave.
#pragma unroll
    for (int off = 8; off >= 1; off >>= 1)
#pragma unroll
        for (int r = 0; r < 8; ++r)
            s2[r] += __shfl_xor(s2[r], off, 16);

    const float b2v = b2[0];
    float wsig[8];
#pragma unroll
    for (int r = 0; r < 8; ++r)
        wsig[r] = 1.0f / (1.0f + __expf(-(s2[r] + b2v)));
    // edge e weight lives in: lane 0 (e<8, reg e) / lane 16 (e>=8, reg e-8)

    // ---- Scatter: agg[tgt] += w * x[src]  (f32 atomics, x is L2-resident)
#pragma unroll 2
    for (int e = 0; e < 16; ++e) {
        int   s  = __shfl(sv, e, 32);
        int   t  = __shfl(sv, 16 + e, 32);
        float wv = __shfl(wsig[e & 7], (e < 8) ? 0 : 16, 32);
        float4 xs = *(const float4*)(x + (size_t)s * H + lane * 4);
        float* dst = agg + (size_t)t * H + lane * 4;
        atomicAdd(dst + 0, wv * xs.x);
        atomicAdd(dst + 1, wv * xs.y);
        atomicAdd(dst + 2, wv * xs.z);
        atomicAdd(dst + 3, wv * xs.w);
    }
}

// ---------------------------------------------------------------------------
// Kernel 4: residual + LayerNorm. One wave per node (H=128 -> 4 floats/lane).
// ---------------------------------------------------------------------------
__global__ void __launch_bounds__(256)
ln_kernel(const float* __restrict__ x, const float* __restrict__ agg,
          const float* __restrict__ gamma, const float* __restrict__ beta,
          float* __restrict__ out)
{
    const int wid  = threadIdx.x >> 5;
    const int lane = threadIdx.x & 31;
    const int node = blockIdx.x * 8 + wid;           // N divisible by 8
    const size_t base = (size_t)node * H + lane * 4;

    float4 xv = *(const float4*)(x + base);
    float4 av = *(const float4*)(agg + base);
    float v0 = xv.x + av.x, v1 = xv.y + av.y;
    float v2 = xv.z + av.z, v3 = xv.w + av.w;

    float sum = v0 + v1 + v2 + v3;
    float ss  = v0 * v0 + v1 * v1 + v2 * v2 + v3 * v3;
#pragma unroll
    for (int off = 16; off >= 1; off >>= 1) {
        sum += __shfl_xor(sum, off, 32);
        ss  += __shfl_xor(ss,  off, 32);
    }
    const float mean = sum * (1.0f / 128.0f);
    const float var  = ss * (1.0f / 128.0f) - mean * mean;
    const float rstd = rsqrtf(var + 1e-5f);

    float4 gv = *(const float4*)(gamma + lane * 4);
    float4 bv = *(const float4*)(beta + lane * 4);
    float4 o;
    o.x = (v0 - mean) * rstd * gv.x + bv.x;
    o.y = (v1 - mean) * rstd * gv.y + bv.y;
    o.z = (v2 - mean) * rstd * gv.z + bv.z;
    o.w = (v3 - mean) * rstd * gv.w + bv.w;
    *(float4*)(out + base) = o;
}

// ---------------------------------------------------------------------------
extern "C" void kernel_launch(void* const* d_in, const int* in_sizes, int n_in,
                              void* d_out, int out_size, void* d_ws, size_t ws_size,
                              hipStream_t stream) {
    const float* x     = (const float*)d_in[0];
    const int*   ei    = (const int*)  d_in[1];
    const float* W1    = (const float*)d_in[2];
    const float* b1    = (const float*)d_in[3];
    const float* W2    = (const float*)d_in[4];
    const float* b2    = (const float*)d_in[5];
    const float* gamma = (const float*)d_in[6];
    const float* beta  = (const float*)d_in[7];
    float* out = (float*)d_out;

    // ws layout: [agg: N*H f32 = 25.6 MB][w1frag: 8*8*32 v16h = 64 KB]
    float* agg    = (float*)d_ws;
    v16h*  w1frag = (v16h*)((char*)d_ws + (size_t)N_NODES * H * sizeof(float));

    zero_agg_kernel<<<(N_NODES * H / 4) / 256, 256, 0, stream>>>((float4*)agg);
    prep_w1_kernel<<<8, 256, 0, stream>>>(W1, w1frag);
    edge_mlp_scatter_kernel<<<N_EDGES / 128, 256, 0, stream>>>(
        x, ei, w1frag, b1, W2, b2, agg);
    ln_kernel<<<N_NODES / 8, 256, 0, stream>>>(x, agg, gamma, beta, out);
}
